// ActivateAttention_49529562858271
// MI455X (gfx1250) — matrix operand order repaired and checked by reading, not compiled, not run on hardware
//
#include <hip/hip_runtime.h>
#include <hip/hip_bf16.h>
#include <math.h>

typedef __attribute__((ext_vector_type(16))) __bf16 v16bf;
typedef __attribute__((ext_vector_type(8)))  __bf16 v8bf;
typedef __attribute__((ext_vector_type(4)))  __bf16 v4bf;
typedef __attribute__((ext_vector_type(8)))  float  v8f;
typedef __attribute__((ext_vector_type(4)))  int    v4i;

#define DIM   768
#define NH    12
#define HD    64
#define NTOK  4096
#define BATCH 2
#define M_TOT (BATCH * NTOK)   // 8192
#define SCALE 0.125f           // 64^-0.5
#define KSTEPS (DIM / 32)      // 24

// ---------------------------------------------------------------------------
// CDNA5 async global->LDS copy (ASYNCcnt path), with sync fallback
// ---------------------------------------------------------------------------
#if __has_builtin(__builtin_amdgcn_global_load_async_to_lds_b128) && \
    __has_builtin(__builtin_amdgcn_s_wait_asynccnt)
#define USE_ASYNC 1
#else
#define USE_ASYNC 0
#endif

__device__ __forceinline__ void copy16_g2l(const __bf16* g, __bf16* l) {
#if USE_ASYNC
  __builtin_amdgcn_global_load_async_to_lds_b128(
      (__attribute__((address_space(1))) v4i*)g,
      (__attribute__((address_space(3))) v4i*)l, 0, 0);
#else
  *(v8bf*)l = *(const v8bf*)g;
#endif
}

__device__ __forceinline__ void wait_async() {
#if USE_ASYNC
  __builtin_amdgcn_s_wait_asynccnt(0);
#endif
}

// ---------------------------------------------------------------------------
// helpers
// ---------------------------------------------------------------------------
__device__ __forceinline__ v8f vzero8() {
  v8f z;
#pragma unroll
  for (int i = 0; i < 8; ++i) z[i] = 0.0f;
  return z;
}

// A-fragment (16x32 bf16): lane m=lane&15, halves hold K {kb..kb+7, kb+16..kb+23},
// kb = (lane>>4)*8.  base -> (row0,k0) of row-major tile, ld in elements.
__device__ __forceinline__ v16bf load_a(const __bf16* base, int ld, int lane) {
  const __bf16* p = base + (lane & 15) * ld + ((lane >> 4) << 3);
  v8bf lo = *(const v8bf*)(p);
  v8bf hi = *(const v8bf*)(p + 16);
  v16bf r;
#pragma unroll
  for (int i = 0; i < 8; ++i) { r[i] = lo[i]; r[i + 8] = hi[i]; }
  return r;
}

// B-fragment (32x16 bf16): lane holds column n=lane&15, K {kb..kb+7, kb+16..kb+23},
// kb = (lane>>4)*8.  base -> row-major [N][K] tile (i.e. B^T).
__device__ __forceinline__ v16bf load_b(const __bf16* base, int ld, int lane) {
  const __bf16* p = base + (lane & 15) * ld + ((lane >> 4) << 3);
  v8bf lo = *(const v8bf*)(p);
  v8bf hi = *(const v8bf*)(p + 16);
  v16bf r;
#pragma unroll
  for (int i = 0; i < 8; ++i) { r[i] = lo[i]; r[i + 8] = hi[i]; }
  return r;
}

__device__ __forceinline__ v8f wmma_bf16(v16bf a, v16bf b, v8f c) {
  return __builtin_amdgcn_wmma_f32_16x16x32_bf16(false, a, false, b, (short)0, c,
                                                 false, false);
}

// stage a 64x32 bf16 tile (row stride DIM in global, 40 in LDS); 128 threads
__device__ __forceinline__ void stage32(const __bf16* g, __bf16* l, int tid) {
#pragma unroll
  for (int i = 0; i < 2; ++i) {
    int c = tid + i * 128;        // 0..255
    int row = c >> 2;             // 0..63
    int col = (c & 3) * 8;        // 0,8,16,24
    copy16_g2l(g + (size_t)row * DIM + col, l + row * 40 + col);
  }
}

// stage a 64x64 bf16 tile (row stride gstride in global, 72 in LDS); 256 threads
__device__ __forceinline__ void stage64(const __bf16* g, size_t gstride,
                                        __bf16* l, int tid) {
#pragma unroll
  for (int i = 0; i < 2; ++i) {
    int c = tid + i * 256;        // 0..511
    int row = c >> 3;             // 0..63
    int col = (c & 7) * 8;        // 0..56
    copy16_g2l(g + (size_t)row * gstride + col, l + row * 72 + col);
  }
}

// ---------------------------------------------------------------------------
// Kernel 0: streaming f32 -> bf16 conversion (activations + weights)
// ---------------------------------------------------------------------------
__global__ __launch_bounds__(256) void cvt_bf16_kernel(
    const float* __restrict__ src, __bf16* __restrict__ dst, int n4) {
  int i = blockIdx.x * 256 + threadIdx.x;
  if (i < n4) {
    float4 v = ((const float4*)src)[i];
    v4bf o;
    o[0] = (__bf16)v.x; o[1] = (__bf16)v.y;
    o[2] = (__bf16)v.z; o[3] = (__bf16)v.w;
    ((v4bf*)dst)[i] = o;
  }
}

// ---------------------------------------------------------------------------
// Kernel 1: QKV projections (bf16 in, WMMA, async double-buffered staging).
// Epilogue: q *= SCALE -> qp[B,H,N,D];  k += bk, exact GELU -> kp[B,H,N,D];
//           v -> vT[B,H,D,N].
// ---------------------------------------------------------------------------
__global__ __launch_bounds__(128) void qkv_proj_kernel(
    const __bf16* __restrict__ xq, const __bf16* __restrict__ xk,
    const __bf16* __restrict__ xv, const __bf16* __restrict__ wqb,
    const __bf16* __restrict__ wkb, const __bf16* __restrict__ wvb,
    const float* __restrict__ bk, __bf16* __restrict__ qp,
    __bf16* __restrict__ kp, __bf16* __restrict__ vT) {
  __shared__ __bf16 Xs[2][64 * 40];
  __shared__ __bf16 Ws[2][64 * 40];
  const int tid  = threadIdx.x;
  const int lane = tid & 31, w = tid >> 5;
  const int mode = blockIdx.z;
  const int m0 = blockIdx.x * 64;
  const int n0 = blockIdx.y * 64;
  const __bf16* X = (mode == 0) ? xq : (mode == 1) ? xk : xv;
  const __bf16* W = (mode == 0) ? wqb : (mode == 1) ? wkb : wvb;
  const __bf16* Xg = X + (size_t)m0 * DIM;
  const __bf16* Wg = W + (size_t)n0 * DIM;

  v8f acc[4];
  for (int i = 0; i < 4; ++i) acc[i] = vzero8();

  stage32(Xg, Xs[0], tid);
  stage32(Wg, Ws[0], tid);
  wait_async();
  __syncthreads();

  for (int ks = 0; ks < KSTEPS; ++ks) {
    const int p = ks & 1;
    if (ks + 1 < KSTEPS) {
      stage32(Xg + (ks + 1) * 32, Xs[p ^ 1], tid);
      stage32(Wg + (ks + 1) * 32, Ws[p ^ 1], tid);
    }
    v16bf a = load_a(&Xs[p][(w * 16) * 40], 40, lane);
    v16bf bfr[4];
#pragma unroll
    for (int nt = 0; nt < 4; ++nt)
      bfr[nt] = load_b(&Ws[p][(nt * 16) * 40], 40, lane);
#pragma unroll
    for (int nt = 0; nt < 4; ++nt) acc[nt] = wmma_bf16(a, bfr[nt], acc[nt]);
    wait_async();
    __syncthreads();
  }

  const int mbase = m0 + w * 16 + ((lane >> 4) << 3);
  const int nl = lane & 15;
#pragma unroll
  for (int nt = 0; nt < 4; ++nt) {
#pragma unroll
    for (int r = 0; r < 8; ++r) {
      float val = acc[nt][r];
      int m = mbase + r;
      int n = n0 + nt * 16 + nl;
      int b = m >> 12, tok = m & (NTOK - 1);
      int h = n >> 6, d = n & (HD - 1);
      if (mode == 0) {
        qp[(((size_t)(b * NH + h)) * NTOK + tok) * HD + d] = (__bf16)(val * SCALE);
      } else if (mode == 1) {
        val += bk[n];
        val = 0.5f * val * (1.0f + erff(val * 0.70710678118654752f));  // exact GELU
        kp[(((size_t)(b * NH + h)) * NTOK + tok) * HD + d] = (__bf16)val;
      } else {
        vT[(((size_t)(b * NH + h)) * HD + d) * NTOK + tok] = (__bf16)val;
      }
    }
  }
}

// ---------------------------------------------------------------------------
// Kernel 2: flash attention.  256 threads (8 waves) per block handle one (b,h)
// and 128 query rows (16 per wave).  Online softmax over K chunks of 64 with
// double-buffered async K/V staging.  Writes x = attn@V as bf16 [B,N,768].
// ---------------------------------------------------------------------------
__global__ __launch_bounds__(256) void flash_attn_kernel(
    const __bf16* __restrict__ qp, const __bf16* __restrict__ kp,
    const __bf16* __restrict__ vT, __bf16* __restrict__ xb) {
  __shared__ __bf16 Ks[2][64 * 72];      // rows = key token, cols = d
  __shared__ __bf16 Vs[2][64 * 72];      // rows = d, cols = key token (V^T)
  __shared__ __bf16 Ps[8 * 16 * 72];     // per-wave P tile for C->A relayout
  const int tid  = threadIdx.x;
  const int lane = tid & 31, w = tid >> 5;
  const int qt = blockIdx.x;   // q tile (128 rows each)
  const int bh = blockIdx.y;   // b*NH + h

  const __bf16* qbase = qp + (((size_t)bh) * NTOK + qt * 128 + w * 16) * HD;
  const __bf16* kbase = kp + ((size_t)bh) * NTOK * HD;
  const __bf16* vbase = vT + ((size_t)bh) * HD * NTOK;

  v16bf aq0 = load_a(qbase, HD, lane);        // d = 0..31  (scale pre-folded)
  v16bf aq1 = load_a(qbase + 32, HD, lane);   // d = 32..63

  v8f o[4];
  for (int i = 0; i < 4; ++i) o[i] = vzero8();
  float m_i[8], l_i[8];
#pragma unroll
  for (int r = 0; r < 8; ++r) { m_i[r] = -INFINITY; l_i[r] = 0.0f; }

  __bf16* pw = &Ps[w * 16 * 72];
  const int half8 = (lane >> 4) << 3;
  const int nl = lane & 15;

  stage64(kbase, HD, Ks[0], tid);
  stage64(vbase, NTOK, Vs[0], tid);
  wait_async();
  __syncthreads();

  for (int kc = 0; kc < NTOK; kc += 64) {
    const int p = (kc >> 6) & 1;
    if (kc + 64 < NTOK) {
      stage64(kbase + (size_t)(kc + 64) * HD, HD, Ks[p ^ 1], tid);
      stage64(vbase + (kc + 64), NTOK, Vs[p ^ 1], tid);
    }

    // S = (q*scale) . K^T  -- contraction over d, two K=32 steps
    v8f s[4];
#pragma unroll
    for (int nt = 0; nt < 4; ++nt) {
      s[nt] = vzero8();
      v16bf b0 = load_b(&Ks[p][(nt * 16) * 72], 72, lane);
      v16bf b1 = load_b(&Ks[p][(nt * 16) * 72 + 32], 72, lane);
      s[nt] = wmma_bf16(aq0, b0, s[nt]);
      s[nt] = wmma_bf16(aq1, b1, s[nt]);
    }

    // online softmax: row M=r(+8) has its 16 N values in one VGPR across a
    // 16-lane half -> shfl_xor 1/2/4/8 reduces exactly within the half
#pragma unroll
    for (int r = 0; r < 8; ++r) {
      float rm = fmaxf(fmaxf(s[0][r], s[1][r]), fmaxf(s[2][r], s[3][r]));
      rm = fmaxf(rm, __shfl_xor(rm, 1));
      rm = fmaxf(rm, __shfl_xor(rm, 2));
      rm = fmaxf(rm, __shfl_xor(rm, 4));
      rm = fmaxf(rm, __shfl_xor(rm, 8));
      float mn = fmaxf(m_i[r], rm);
      float alpha = __expf(m_i[r] - mn);
      m_i[r] = mn;
      float rs = 0.0f;
#pragma unroll
      for (int nt = 0; nt < 4; ++nt) {
        float pv = __expf(s[nt][r] - mn);
        s[nt][r] = pv;
        rs += pv;
      }
      rs += __shfl_xor(rs, 1);
      rs += __shfl_xor(rs, 2);
      rs += __shfl_xor(rs, 4);
      rs += __shfl_xor(rs, 8);
      l_i[r] = l_i[r] * alpha + rs;
#pragma unroll
      for (int dt = 0; dt < 4; ++dt) o[dt][r] *= alpha;
    }

    // P: C layout -> bf16 A layout via per-wave LDS tile
#pragma unroll
    for (int nt = 0; nt < 4; ++nt)
#pragma unroll
      for (int r = 0; r < 8; ++r)
        pw[(r + half8) * 72 + nt * 16 + nl] = (__bf16)s[nt][r];

    v16bf pa0 = load_a(pw, 72, lane);        // k = 0..31 of chunk
    v16bf pa1 = load_a(pw + 32, 72, lane);   // k = 32..63

    // O += P . V  (B columns = d, contiguous K thanks to V^T layout)
#pragma unroll
    for (int dt = 0; dt < 4; ++dt) {
      v16bf b0 = load_b(&Vs[p][(dt * 16) * 72], 72, lane);
      v16bf b1 = load_b(&Vs[p][(dt * 16) * 72 + 32], 72, lane);
      o[dt] = wmma_bf16(pa0, b0, o[dt]);
      o[dt] = wmma_bf16(pa1, b1, o[dt]);
    }

    wait_async();
    __syncthreads();
  }

  const int b = bh / NH, h = bh % NH;
#pragma unroll
  for (int r = 0; r < 8; ++r) {
    float inv = 1.0f / l_i[r];
    int tok = qt * 128 + w * 16 + half8 + r;
#pragma unroll
    for (int dt = 0; dt < 4; ++dt) {
      int feat = h * HD + dt * 16 + nl;
      xb[((size_t)(b * NTOK + tok)) * DIM + feat] = (__bf16)(o[dt][r] * inv);
    }
  }
}

// ---------------------------------------------------------------------------
// Kernel 3: output projection  out = x @ Wp^T + bp   (f32 output)
// ---------------------------------------------------------------------------
__global__ __launch_bounds__(128) void out_proj_kernel(
    const __bf16* __restrict__ xb, const __bf16* __restrict__ wpb,
    const float* __restrict__ bp, float* __restrict__ out) {
  __shared__ __bf16 Xs[2][64 * 40];
  __shared__ __bf16 Ws[2][64 * 40];
  const int tid  = threadIdx.x;
  const int lane = tid & 31, w = tid >> 5;
  const int m0 = blockIdx.x * 64;
  const int n0 = blockIdx.y * 64;
  const __bf16* Xg = xb + (size_t)m0 * DIM;
  const __bf16* Wg = wpb + (size_t)n0 * DIM;

  v8f acc[4];
  for (int i = 0; i < 4; ++i) acc[i] = vzero8();

  stage32(Xg, Xs[0], tid);
  stage32(Wg, Ws[0], tid);
  wait_async();
  __syncthreads();

  for (int ks = 0; ks < KSTEPS; ++ks) {
    const int p = ks & 1;
    if (ks + 1 < KSTEPS) {
      stage32(Xg + (ks + 1) * 32, Xs[p ^ 1], tid);
      stage32(Wg + (ks + 1) * 32, Ws[p ^ 1], tid);
    }
    v16bf a = load_a(&Xs[p][(w * 16) * 40], 40, lane);
    v16bf bfr[4];
#pragma unroll
    for (int nt = 0; nt < 4; ++nt)
      bfr[nt] = load_b(&Ws[p][(nt * 16) * 40], 40, lane);
#pragma unroll
    for (int nt = 0; nt < 4; ++nt) acc[nt] = wmma_bf16(a, bfr[nt], acc[nt]);
    wait_async();
    __syncthreads();
  }

  const int mbase = m0 + w * 16 + ((lane >> 4) << 3);
  const int nl = lane & 15;
#pragma unroll
  for (int nt = 0; nt < 4; ++nt) {
#pragma unroll
    for (int r = 0; r < 8; ++r) {
      int m = mbase + r;
      int n = n0 + nt * 16 + nl;
      out[(size_t)m * DIM + n] = acc[nt][r] + bp[n];
    }
  }
}

// ---------------------------------------------------------------------------
// launch
// ---------------------------------------------------------------------------
extern "C" void kernel_launch(void* const* d_in, const int* in_sizes, int n_in,
                              void* d_out, int out_size, void* d_ws,
                              size_t ws_size, hipStream_t stream) {
  (void)in_sizes; (void)n_in; (void)out_size; (void)ws_size;
  const float* query = (const float*)d_in[0];
  const float* key   = (const float*)d_in[1];
  const float* value = (const float*)d_in[2];
  const float* Wq    = (const float*)d_in[3];
  const float* Wk    = (const float*)d_in[4];
  const float* bk    = (const float*)d_in[5];
  const float* Wv    = (const float*)d_in[6];
  const float* Wp    = (const float*)d_in[7];
  const float* bp    = (const float*)d_in[8];

  const size_t ACT = (size_t)M_TOT * DIM;   // 6,291,456 elements
  const size_t WTS = (size_t)DIM * DIM;     //   589,824 elements
  __bf16* base = (__bf16*)d_ws;
  __bf16* xq  = base;            // [8192, 768] bf16
  __bf16* xk  = xq + ACT;
  __bf16* xv  = xk + ACT;
  __bf16* wqb = xv + ACT;        // [768, 768] bf16 x4
  __bf16* wkb = wqb + WTS;
  __bf16* wvb = wkb + WTS;
  __bf16* wpb = wvb + WTS;
  __bf16* qp  = wpb + WTS;       // [B,H,N,D] (pre-scaled)
  __bf16* kp  = qp + ACT;        // [B,H,N,D] (biased + gelu)
  __bf16* vT  = kp + ACT;        // [B,H,D,N]
  __bf16* xb  = vT + ACT;        // [B,N,768]

  const int ACT4 = (int)(ACT / 4), WTS4 = (int)(WTS / 4);
  cvt_bf16_kernel<<<(ACT4 + 255) / 256, 256, 0, stream>>>(query, xq, ACT4);
  cvt_bf16_kernel<<<(ACT4 + 255) / 256, 256, 0, stream>>>(key, xk, ACT4);
  cvt_bf16_kernel<<<(ACT4 + 255) / 256, 256, 0, stream>>>(value, xv, ACT4);
  cvt_bf16_kernel<<<(WTS4 + 255) / 256, 256, 0, stream>>>(Wq, wqb, WTS4);
  cvt_bf16_kernel<<<(WTS4 + 255) / 256, 256, 0, stream>>>(Wk, wkb, WTS4);
  cvt_bf16_kernel<<<(WTS4 + 255) / 256, 256, 0, stream>>>(Wv, wvb, WTS4);
  cvt_bf16_kernel<<<(WTS4 + 255) / 256, 256, 0, stream>>>(Wp, wpb, WTS4);

  qkv_proj_kernel<<<dim3(M_TOT / 64, DIM / 64, 3), 128, 0, stream>>>(
      xq, xk, xv, wqb, wkb, wvb, bk, qp, kp, vT);
  flash_attn_kernel<<<dim3(NTOK / 128, BATCH * NH), 256, 0, stream>>>(
      qp, kp, vT, xb);
  out_proj_kernel<<<dim3(M_TOT / 64, DIM / 64), 128, 0, stream>>>(
      xb, wpb, bp, (float*)d_out);
}
